// Lag_Net_R1_T1_56521769615684
// MI455X (gfx1250) — matrix operand order repaired and checked by reading, not compiled, not run on hardware
//
#include <hip/hip_runtime.h>
#include <hip/hip_bf16.h>

typedef __attribute__((ext_vector_type(16))) _Float16 v16h;
typedef __attribute__((ext_vector_type(8)))  _Float16 v8h;
typedef __attribute__((ext_vector_type(8)))  float    v8f;
typedef __attribute__((ext_vector_type(4)))  float    vf4;

union Frag { v16h v; _Float16 e[16]; };

#define H 256
#define WAVES_PER_BLOCK 4
#define ROWS_PER_WAVE 16

// Branchless tanh: use gfx1250 hardware v_tanh_f32 when clang exposes it,
// else exp2+rcp (both single-issue TRANS/VALU ops, no EXEC manipulation).
__device__ __forceinline__ float fast_tanh(float x)
{
#if __has_builtin(__builtin_amdgcn_tanhf)
    return __builtin_amdgcn_tanhf(x);
#elif __has_builtin(__builtin_amdgcn_tanh_f32)
    return __builtin_amdgcn_tanh_f32(x);
#else
    // tanh(x) = 1 - 2/(exp2(2*log2e*x) + 1); saturates correctly at +-inf
    float e = __builtin_amdgcn_exp2f(x * 2.8853900817779268f);
    return 1.0f - 2.0f * __builtin_amdgcn_rcpf(e + 1.0f);
#endif
}

// Load 8 consecutive floats as two float4s (coalesced b128 pairs).
struct F8 { float v[8]; };
__device__ __forceinline__ F8 ld8(const float* __restrict__ p)
{
    F8 r;
    vf4 a = *(const vf4*)p;
    vf4 b = *(const vf4*)(p + 4);
#pragma unroll
    for (int j = 0; j < 4; ++j) { r.v[j] = a[j]; r.v[4 + j] = b[j]; }
    return r;
}

// ---------------------------------------------------------------------------
// Prep: convert the three 256x256 f32 weight matrices (plus transposes for
// the two backward GEMMs) to f16 in *fragment-tiled* layout:
//   tile t = nt*8 + k  (nt: 16-col block, k: 32-row block)
//   halves[ t*512 + lane*16 + h ] = W[(k*32 + 16*(h>>3) + 8*(lane>>4) + (h&7))*256 + nt*16 + (lane&15)]
// so each WMMA B-fragment is one contiguous lane-coalesced 1KB block.
// ws layout (halves): [0]=Mw2, [65536]=Mw2T, [131072]=Vw2, [196608]=Vw2T,
// [262144]=Gw2.  Total 640 KB.
// ---------------------------------------------------------------------------
__global__ void prep_weights(const float* __restrict__ Mw2,
                             const float* __restrict__ Vw2,
                             const float* __restrict__ Gw2,
                             _Float16* __restrict__ ws)
{
    int i = blockIdx.x * blockDim.x + threadIdx.x;   // 0..65535
    if (i < H * H) {
        const int h  = i & 15;
        const int kl = (i >> 4) & 31;
        const int t  = i >> 9;          // nt*8 + k
        const int k  = t & 7, nt = t >> 3;
        const int srow = k * 32 + ((h & 8) << 1) + ((kl >> 4) << 3) + (h & 7);   // K index
        const int scol = nt * 16 + (kl & 15);   // N index
        ws[i]            = (_Float16)Mw2[srow * H + scol];
        ws[65536  + i]   = (_Float16)Mw2[scol * H + srow];
        ws[131072 + i]   = (_Float16)Vw2[srow * H + scol];
        ws[196608 + i]   = (_Float16)Vw2[scol * H + srow];
        ws[262144 + i]   = (_Float16)Gw2[srow * H + scol];
    }
}

// One 16x256 = A(16x256,f16-regs) @ W(256x256,f16-tiled-global) GEMM pass.
__device__ __forceinline__ void gemm_tile(const Frag* A,
                                          const _Float16* __restrict__ W,
                                          const float* __restrict__ b2,
                                          bool act, int lane,
                                          _Float16* __restrict__ O)
{
    const int row16 = lane & 15;
    const int orow0 = (lane & 16) ? 8 : 0;
    const int loff  = lane * 16;
#pragma unroll 1
    for (int nt = 0; nt < 16; ++nt) {
        const int nc0 = nt * 16;
        float bias = b2 ? b2[nc0 + row16] : 0.0f;
        v8f acc;
#pragma unroll
        for (int r = 0; r < 8; ++r) acc[r] = bias;
#pragma unroll
        for (int k = 0; k < 8; ++k) {
            v16h b = *(const v16h*)(W + (size_t)(((nt * 8 + k) << 9) + loff));
            acc = __builtin_amdgcn_wmma_f32_16x16x32_f16(
                false, A[k].v, false, b, (short)0, acc, false, false);
        }
        const int ocol = nc0 + row16;
#pragma unroll
        for (int r = 0; r < 8; ++r) {
            float v = acc[r];
            if (act) v = fast_tanh(v);
            O[(orow0 + r) * H + ocol] = (_Float16)v;
        }
    }
}

__device__ __forceinline__ float combine_half(float v)
{
    return v + __shfl_down(v, 16, 32);   // lane L += lane L+16 (wave32)
}

// ---------------------------------------------------------------------------
// Main kernel: each wave handles 16 batch rows. Lanes L and L+16 both own row
// (L&15); they split the 256 hidden units per the WMMA A-fragment K layout:
//   unit = 32*f + 16*g + j + koff,  g in {0,1}, j in 0..7,
//   fragment half index h = 8*g + j,  koff = 8 for lanes >= 16
// so layer-1 outputs are produced directly as register-resident A fragments,
// and every per-unit loop touches 8 consecutive units (vectorizable loads).
// ---------------------------------------------------------------------------
extern "C" __global__ void __launch_bounds__(128, 1)
lagnet_main(const float* __restrict__ x,
            const float* __restrict__ Mw1, const float* __restrict__ Mb1,
            const float* __restrict__ Mb2, const float* __restrict__ Mw3,
            const float* __restrict__ Mb3,
            const float* __restrict__ Vw1, const float* __restrict__ Vb1,
            const float* __restrict__ Vb2, const float* __restrict__ Vw3,
            const float* __restrict__ Gw1, const float* __restrict__ Gb1,
            const float* __restrict__ Gb2, const float* __restrict__ Gw3,
            const float* __restrict__ Gb3,
            const _Float16* __restrict__ Mw2h, const _Float16* __restrict__ Mw2Th,
            const _Float16* __restrict__ Vw2h, const _Float16* __restrict__ Vw2Th,
            const _Float16* __restrict__ Gw2h,
            float* __restrict__ out)
{
    extern __shared__ _Float16 smem[];           // 16KB per wave (2 tiles)
    const int lane  = threadIdx.x & 31;
    const int wave  = threadIdx.x >> 5;
    _Float16* O1 = smem + wave * 8192;           // 16x256 f16
    _Float16* O2 = O1 + 4096;                    // 16x256 f16

    const int row16 = lane & 15;
    const int koff  = (lane & 16) ? 8 : 0;
    const long long row =
        (long long)(blockIdx.x * WAVES_PER_BLOCK + wave) * ROWS_PER_WAVE + row16;

    // ---- load input row (lanes L and L+16 both load row L&15) -------------
    const float* xr = x + row * 7;
    const float q0 = xr[0], q1 = xr[1], q2 = xr[2];
    const float rdot = xr[3], phidot = xr[4], u0 = xr[5], u1 = xr[6];
    const float cosphi = q1, sinphi = q2;
    const float dcos = -sinphi * phidot, dsin = cosphi * phidot;
    const float d0 = rdot, d1 = dcos, d2 = dsin;            // q3_dot
    const float c0 = rdot * rdot, c1 = rdot * phidot;       // qdot (x) qdot
    const float c2 = c1, c3 = phidot * phidot;

    Frag AH[8], AT[8];
    float Mq[4] = {0,0,0,0}, dM[4] = {0,0,0,0}, gq[4] = {0,0,0,0};
    float dS3[3] = {0,0,0}, dV3[3] = {0,0,0};

    // ---- P1: M layer-1 primal + JVP tangent, straight into A fragments ----
#pragma unroll
    for (int f = 0; f < 8; ++f) {
#pragma unroll
        for (int g = 0; g < 2; ++g) {
            const int ub = 32 * f + 16 * g + koff;
            F8 w0 = ld8(Mw1 + ub), w1 = ld8(Mw1 + H + ub);
            F8 w2 = ld8(Mw1 + 2 * H + ub), bb = ld8(Mb1 + ub);
#pragma unroll
            for (int j = 0; j < 8; ++j) {
                const float z  = q0 * w0.v[j] + q1 * w1.v[j] + q2 * w2.v[j] + bb.v[j];
                const float hv = fast_tanh(z);
                const float tv = (d0 * w0.v[j] + d1 * w1.v[j] + d2 * w2.v[j])
                                 * (1.0f - hv * hv);
                AH[f].e[8 * g + j] = (_Float16)hv;
                AT[f].e[8 * g + j] = (_Float16)tv;
            }
        }
    }

    // ---- P2: fused dual GEMM (primal + tangent share B fragments) ---------
    {
        const int orow0 = (lane & 16) ? 8 : 0;
        const int loff  = lane * 16;
#pragma unroll 1
        for (int nt = 0; nt < 16; ++nt) {
            const int nc0 = nt * 16;
            const float bias = Mb2[nc0 + row16];
            v8f accH, accT;
#pragma unroll
            for (int r = 0; r < 8; ++r) { accH[r] = bias; accT[r] = 0.0f; }
#pragma unroll
            for (int k = 0; k < 8; ++k) {
                v16h b = *(const v16h*)(Mw2h + (size_t)(((nt * 8 + k) << 9) + loff));
                accH = __builtin_amdgcn_wmma_f32_16x16x32_f16(
                    false, AH[k].v, false, b, (short)0, accH, false, false);
                accT = __builtin_amdgcn_wmma_f32_16x16x32_f16(
                    false, AT[k].v, false, b, (short)0, accT, false, false);
            }
            const int ocol = nc0 + row16;
#pragma unroll
            for (int r = 0; r < 8; ++r) {
                const float hv = fast_tanh(accH[r]);
                O1[(orow0 + r) * H + ocol] = (_Float16)hv;                         // h2
                O2[(orow0 + r) * H + ocol] = (_Float16)((1.0f - hv*hv) * accT[r]); // t2
            }
        }
    }

    // ---- P3: layer-3 M/dM dots, build d2M fragment for backward -----------
#pragma unroll
    for (int f = 0; f < 8; ++f) {
#pragma unroll
        for (int g = 0; g < 2; ++g) {
            const int ub = 32 * f + 16 * g + koff;
            v8h h2v = *(const v8h*)(O1 + row16 * H + ub);
            v8h t2v = *(const v8h*)(O2 + row16 * H + ub);
#pragma unroll
            for (int j = 0; j < 8; ++j) {
                const int u = ub + j;
                const float h2 = (float)h2v[j];
                const float t2 = (float)t2v[j];
                vf4 w3 = *(const vf4*)(Mw3 + (size_t)u * 4);
                Mq[0] += h2*w3[0]; Mq[1] += h2*w3[1]; Mq[2] += h2*w3[2]; Mq[3] += h2*w3[3];
                dM[0] += t2*w3[0]; dM[1] += t2*w3[1]; dM[2] += t2*w3[2]; dM[3] += t2*w3[3];
                const float e  = 1.0f - h2 * h2;
                const float cd = c0*w3[0] + c1*w3[1] + c2*w3[2] + c3*w3[3];  // c @ Mw3^T
                AT[f].e[8 * g + j] = (_Float16)(cd * e);                     // d2M
            }
        }
    }

    // ---- P4: g1M = d2M @ Mw2^T --------------------------------------------
    gemm_tile(AT, Mw2Th, nullptr, false, lane, O1);

    // ---- P5: dS = (g1M * sech^2(z1)) @ Mw1^T ------------------------------
#pragma unroll
    for (int f = 0; f < 8; ++f) {
#pragma unroll
        for (int g = 0; g < 2; ++g) {
            const int ub = 32 * f + 16 * g + koff;
            v8h gv = *(const v8h*)(O1 + row16 * H + ub);
            F8 w0 = ld8(Mw1 + ub), w1 = ld8(Mw1 + H + ub), w2 = ld8(Mw1 + 2 * H + ub);
#pragma unroll
            for (int j = 0; j < 8; ++j) {
                const float h1 = (float)AH[f].e[8 * g + j];
                const float gp = (float)gv[j] * (1.0f - h1 * h1);
                dS3[0] += gp * w0.v[j];
                dS3[1] += gp * w1.v[j];
                dS3[2] += gp * w2.v[j];
            }
        }
    }

    // ---- P6: V layer-1 -> fragments, layer-2 GEMM -------------------------
#pragma unroll
    for (int f = 0; f < 8; ++f) {
#pragma unroll
        for (int g = 0; g < 2; ++g) {
            const int ub = 32 * f + 16 * g + koff;
            F8 w0 = ld8(Vw1 + ub), w1 = ld8(Vw1 + H + ub);
            F8 w2 = ld8(Vw1 + 2 * H + ub), bb = ld8(Vb1 + ub);
#pragma unroll
            for (int j = 0; j < 8; ++j) {
                const float z = q0*w0.v[j] + q1*w1.v[j] + q2*w2.v[j] + bb.v[j];
                AH[f].e[8 * g + j] = (_Float16)fast_tanh(z);
            }
        }
    }
    gemm_tile(AH, Vw2h, Vb2, true, lane, O1);     // a2 = tanh(a1 @ Vw2 + b2)

    // ---- P7: d2V = Vw3 * sech^2(z2) ---------------------------------------
#pragma unroll
    for (int f = 0; f < 8; ++f) {
#pragma unroll
        for (int g = 0; g < 2; ++g) {
            const int ub = 32 * f + 16 * g + koff;
            v8h a2v = *(const v8h*)(O1 + row16 * H + ub);
            F8 w3 = ld8(Vw3 + ub);
#pragma unroll
            for (int j = 0; j < 8; ++j) {
                const float a2 = (float)a2v[j];
                AT[f].e[8 * g + j] = (_Float16)(w3.v[j] * (1.0f - a2 * a2));
            }
        }
    }

    // ---- P8: g1V = d2V @ Vw2^T --------------------------------------------
    gemm_tile(AT, Vw2Th, nullptr, false, lane, O1);

    // ---- P9: dV = (g1V * sech^2(z1)) @ Vw1^T ------------------------------
#pragma unroll
    for (int f = 0; f < 8; ++f) {
#pragma unroll
        for (int g = 0; g < 2; ++g) {
            const int ub = 32 * f + 16 * g + koff;
            v8h gv = *(const v8h*)(O1 + row16 * H + ub);
            F8 w0 = ld8(Vw1 + ub), w1 = ld8(Vw1 + H + ub), w2 = ld8(Vw1 + 2 * H + ub);
#pragma unroll
            for (int j = 0; j < 8; ++j) {
                const float a1 = (float)AH[f].e[8 * g + j];
                const float gp = (float)gv[j] * (1.0f - a1 * a1);
                dV3[0] += gp * w0.v[j];
                dV3[1] += gp * w1.v[j];
                dV3[2] += gp * w2.v[j];
            }
        }
    }

    // ---- P10: G forward ----------------------------------------------------
#pragma unroll
    for (int f = 0; f < 8; ++f) {
#pragma unroll
        for (int g = 0; g < 2; ++g) {
            const int ub = 32 * f + 16 * g + koff;
            F8 w0 = ld8(Gw1 + ub), w1 = ld8(Gw1 + H + ub);
            F8 w2 = ld8(Gw1 + 2 * H + ub), bb = ld8(Gb1 + ub);
#pragma unroll
            for (int j = 0; j < 8; ++j) {
                const float z = q0*w0.v[j] + q1*w1.v[j] + q2*w2.v[j] + bb.v[j];
                AH[f].e[8 * g + j] = (_Float16)fast_tanh(z);
            }
        }
    }
    gemm_tile(AH, Gw2h, Gb2, true, lane, O1);

    // ---- P11: g_q = h2G @ Gw3 ---------------------------------------------
#pragma unroll
    for (int f = 0; f < 8; ++f) {
#pragma unroll
        for (int g = 0; g < 2; ++g) {
            const int ub = 32 * f + 16 * g + koff;
            v8h hv = *(const v8h*)(O1 + row16 * H + ub);
#pragma unroll
            for (int j = 0; j < 8; ++j) {
                const float hg = (float)hv[j];
                vf4 w3 = *(const vf4*)(Gw3 + (size_t)(ub + j) * 4);
                gq[0] += hg*w3[0]; gq[1] += hg*w3[1];
                gq[2] += hg*w3[2]; gq[3] += hg*w3[3];
            }
        }
    }

    // ---- P12: combine lane-halves, 2x2 solve, store -----------------------
#pragma unroll
    for (int i = 0; i < 4; ++i) {
        Mq[i] = combine_half(Mq[i]);
        dM[i] = combine_half(dM[i]);
        gq[i] = combine_half(gq[i]);
    }
#pragma unroll
    for (int i = 0; i < 3; ++i) {
        dS3[i] = combine_half(dS3[i]);
        dV3[i] = combine_half(dV3[i]);
    }

    if (lane < 16) {
#pragma unroll
        for (int i = 0; i < 4; ++i) { Mq[i] += Mb3[i]; gq[i] += Gb3[i]; }
        const float dVq0 = dV3[0];
        const float dVq1 = dV3[1] * (-sinphi) + dV3[2] * cosphi;
        const float dSq0 = dS3[0];
        const float dSq1 = dS3[1] * (-sinphi) + dS3[2] * cosphi;
        const float t0 = -(dM[0]*rdot + dM[1]*phidot) + 0.5f*dSq0 - dVq0
                         + gq[0]*u0 + gq[1]*u1;
        const float t1 = -(dM[2]*rdot + dM[3]*phidot) + 0.5f*dSq1 - dVq1
                         + gq[2]*u0 + gq[3]*u1;
        const float det = Mq[0]*Mq[3] - Mq[1]*Mq[2];
        const float inv = 1.0f / det;
        const float dqd0 = ( Mq[3]*t0 - Mq[1]*t1) * inv;
        const float dqd1 = (-Mq[2]*t0 + Mq[0]*t1) * inv;
        float* o = out + row * 7;
        o[0] = rdot; o[1] = dcos; o[2] = dsin;
        o[3] = dqd0; o[4] = dqd1; o[5] = 0.0f; o[6] = 0.0f;
    }
}

// ---------------------------------------------------------------------------
extern "C" void kernel_launch(void* const* d_in, const int* in_sizes, int n_in,
                              void* d_out, int out_size, void* d_ws, size_t ws_size,
                              hipStream_t stream)
{
    (void)in_sizes; (void)n_in; (void)out_size; (void)ws_size;
    const float* x   = (const float*)d_in[1];
    const float* Mw1 = (const float*)d_in[2];
    const float* Mb1 = (const float*)d_in[3];
    const float* Mw2 = (const float*)d_in[4];
    const float* Mb2 = (const float*)d_in[5];
    const float* Mw3 = (const float*)d_in[6];
    const float* Mb3 = (const float*)d_in[7];
    const float* Vw1 = (const float*)d_in[8];
    const float* Vb1 = (const float*)d_in[9];
    const float* Vw2 = (const float*)d_in[10];
    const float* Vb2 = (const float*)d_in[11];
    const float* Vw3 = (const float*)d_in[12];
    const float* Gw1 = (const float*)d_in[14];
    const float* Gb1 = (const float*)d_in[15];
    const float* Gw2 = (const float*)d_in[16];
    const float* Gb2 = (const float*)d_in[17];
    const float* Gw3 = (const float*)d_in[18];
    const float* Gb3 = (const float*)d_in[19];

    _Float16* wh = (_Float16*)d_ws;

    prep_weights<<<256, 256, 0, stream>>>(Mw2, Vw2, Gw2, wh);

    const int B = 131072;
    const int rows_per_block = WAVES_PER_BLOCK * ROWS_PER_WAVE;   // 64
    lagnet_main<<<B / rows_per_block, WAVES_PER_BLOCK * 32, 64 * 1024, stream>>>(
        x, Mw1, Mb1, Mb2, Mw3, Mb3,
        Vw1, Vb1, Vb2, Vw3,
        Gw1, Gb1, Gb2, Gw3, Gb3,
        wh, wh + 65536, wh + 131072, wh + 196608, wh + 262144,
        (float*)d_out);
}
